// Model_52518860096592
// MI455X (gfx1250) — compile-verified, matrix operand order checked by reading, not run on hardware
//
#include <hip/hip_runtime.h>

// ---------------------------------------------------------------------------
// Flash attention: B=4, Sq=Skv=4096, D=64, fp32 in/out.
// CDNA5 (gfx1250) wave32, v_wmma_f32_16x16x32_bf16.
//  - double-buffered LDS tiles, single barrier per KV iteration
//  - DPP-based softmax row max (no LDS permutes)
//  - softmax row-sum fused into WMMA via an all-ones B fragment
//  - log2-domain softmax (bare v_exp_f32)
// ---------------------------------------------------------------------------

typedef __attribute__((ext_vector_type(16))) __bf16 v16bf;
typedef __attribute__((ext_vector_type(8)))  __bf16 v8bf;
typedef __attribute__((ext_vector_type(4)))  __bf16 v4bf;
typedef __attribute__((ext_vector_type(8)))  float  v8f;
typedef __attribute__((ext_vector_type(4)))  float  v4f;

#define ATT_B   4
#define ATT_SQ  4096
#define ATT_SKV 4096
#define ATT_D   64
#define BLK_M   64    // queries per workgroup (4 waves x 16 rows)
#define BLK_N   64    // kv keys per iteration
#define NWAVES  4
#define KPAD    72    // padded K row (bf16) -> conflict-free ds_load_b128
#define VPAD    72    // padded V^T row
#define PPAD    72    // padded P row

#define LOG2E      1.44269504088896340736f
#define SCALE_LOG2 (0.125f * LOG2E)

__device__ __forceinline__ __bf16 f2bf(float f) {
  unsigned u = __builtin_bit_cast(unsigned, f);
  unsigned r = u + 0x7FFFu + ((u >> 16) & 1u);   // round-to-nearest-even
  unsigned short h = (unsigned short)(r >> 16);
  return __builtin_bit_cast(__bf16, h);
}

__device__ __forceinline__ v8f zero8() {
  v8f z;
#pragma unroll
  for (int i = 0; i < 8; ++i) z[i] = 0.0f;
  return z;
}

// Butterfly max across a 16-lane row: xor 1/2/4 via DPP8, cross-half via
// DPP16 ROW_MIRROR (0x140). Pure VALU -- no LDS traffic, no DScnt waits.
template <int K>
__device__ __forceinline__ float dpp8_xor_max(float v) {
  constexpr int sel = ((0 ^ K) & 7)        | (((1 ^ K) & 7) << 3)  |
                      (((2 ^ K) & 7) << 6) | (((3 ^ K) & 7) << 9)  |
                      (((4 ^ K) & 7) << 12)| (((5 ^ K) & 7) << 15) |
                      (((6 ^ K) & 7) << 18)| (((7 ^ K) & 7) << 21);
  int t = __builtin_amdgcn_mov_dpp8(__builtin_bit_cast(int, v), sel);
  return fmaxf(v, __builtin_bit_cast(float, t));
}

__device__ __forceinline__ float rowmax16(float v) {
  v = dpp8_xor_max<1>(v);
  v = dpp8_xor_max<2>(v);
  v = dpp8_xor_max<4>(v);
  int t = __builtin_amdgcn_update_dpp(0, __builtin_bit_cast(int, v),
                                      0x140 /*ROW_MIRROR*/, 0xf, 0xf, true);
  return fmaxf(v, __builtin_bit_cast(float, t));
}

__global__ __launch_bounds__(128, 2)
void fa_fwd_gfx1250(const float* __restrict__ Q, const float* __restrict__ K,
                    const float* __restrict__ V, const float* __restrict__ Msk,
                    float* __restrict__ O) {
  __shared__ alignas(16) __bf16 Klds[2][BLK_N * KPAD];         // [buf][key][d]
  __shared__ alignas(16) __bf16 Vt[2][ATT_D * VPAD];           // [buf][d][key]
  __shared__ alignas(16) __bf16 Plds[NWAVES][16 * PPAD];       // per-wave P

  const int b    = blockIdx.y;
  const int q0   = blockIdx.x * BLK_M;
  const int tid  = threadIdx.x;
  const int wave = tid >> 5;
  const int lane = tid & 31;
  const int ln   = lane & 15;
  const int half = lane >> 4;

  const float* Qb = Q + (size_t)b * ATT_SQ  * ATT_D;
  const float* Kb = K + (size_t)b * ATT_SKV * ATT_D;
  const float* Vb = V + (size_t)b * ATT_SKV * ATT_D;
  const float* Mb = Msk + (size_t)b * ATT_SKV;
  float*       Ob = O + (size_t)b * ATT_SQ  * ATT_D;

  // ---- Q fragments (A-matrix 16x32 bf16; d = kk*32 + (j>>3)*16 + half*8 + (j&7))
  const int qrow = q0 + wave * 16 + ln;
  v16bf qf[2];
#pragma unroll
  for (int kk = 0; kk < 2; ++kk) {
#pragma unroll
    for (int g = 0; g < 2; ++g) {
      const int d0 = kk * 32 + g * 16 + half * 8;
      const v4f a = *(const v4f*)(Qb + (size_t)qrow * ATT_D + d0);
      const v4f c = *(const v4f*)(Qb + (size_t)qrow * ATT_D + d0 + 4);
#pragma unroll
      for (int t = 0; t < 4; ++t) {
        qf[kk][g * 8 + t]     = f2bf(a[t]);
        qf[kk][g * 8 + 4 + t] = f2bf(c[t]);
      }
    }
  }

  // All-ones B fragment: accL = wmma(P, ones, accL) accumulates softmax row sums.
  v16bf ones;
#pragma unroll
  for (int j = 0; j < 16; ++j)
    ones[j] = __builtin_bit_cast(__bf16, (unsigned short)0x3F80);

  v8f acc[4];
#pragma unroll
  for (int i = 0; i < 4; ++i) acc[i] = zero8();
  v8f accL = zero8();
  float mrow[8];
#pragma unroll
  for (int r = 0; r < 8; ++r) mrow[r] = -1.0e30f;

  const int skey = tid >> 4;           // staging key row (advances by 8 per i)
  const int sd   = (tid & 15) * 4;     // staging d offset

  // ---- prologue: stage tile 0 into buffer 0
#pragma unroll
  for (int i = 0; i < 8; ++i) {
    const int key = skey + i * 8;
    const v4f kd = *(const v4f*)(Kb + (size_t)key * ATT_D + sd);
    const v4f vd = *(const v4f*)(Vb + (size_t)key * ATT_D + sd);
    v4bf kp;
#pragma unroll
    for (int t = 0; t < 4; ++t) kp[t] = f2bf(kd[t]);
    *(v4bf*)&Klds[0][key * KPAD + sd] = kp;
#pragma unroll
    for (int t = 0; t < 4; ++t) Vt[0][(sd + t) * VPAD + key] = f2bf(vd[t]);
  }
  __syncthreads();

  int p = 0;
  for (int kv0 = 0; kv0 < ATT_SKV; kv0 += BLK_N, p ^= 1) {
    const bool has_next = (kv0 + BLK_N) < ATT_SKV;

    // ---- issue global loads for the NEXT tile now; they complete during
    //      this tile's GEMM/softmax work (single-barrier double buffering)
    v4f nkd[8], nvd[8];
    if (has_next) {
#pragma unroll
      for (int i = 0; i < 8; ++i) {
        const int key = kv0 + BLK_N + skey + i * 8;
        nkd[i] = *(const v4f*)(Kb + (size_t)key * ATT_D + sd);
        nvd[i] = *(const v4f*)(Vb + (size_t)key * ATT_D + sd);
      }
      if (kv0 + 2 * BLK_N < ATT_SKV) {
        __builtin_prefetch(Kb + (size_t)(kv0 + 2 * BLK_N + skey) * ATT_D + sd, 0, 3);
        __builtin_prefetch(Vb + (size_t)(kv0 + 2 * BLK_N + skey) * ATT_D + sd, 0, 3);
      }
    }

    // ---- S = Q * K^T : four 16x16 N-tiles, k=64 via two wmma k-steps each
    v8f s[4];
#pragma unroll
    for (int n = 0; n < 4; ++n) {
      v8f z = zero8();
#pragma unroll
      for (int kk = 0; kk < 2; ++kk) {
        const int row = n * 16 + ln;
        const int d0  = kk * 32 + half * 8;
        const v8bf lo = *(const v8bf*)&Klds[p][row * KPAD + d0];
        const v8bf hi = *(const v8bf*)&Klds[p][row * KPAD + d0 + 16];
        v16bf kb;
#pragma unroll
        for (int t = 0; t < 8; ++t) { kb[t] = lo[t]; kb[8 + t] = hi[t]; }
        z = __builtin_amdgcn_wmma_f32_16x16x32_bf16(false, qf[kk], false, kb,
                                                    (short)0, z, false, false);
      }
      s[n] = z;
    }

    // ---- scale+mask in log2 domain: s2 = s*(scale*log2e) + mask*log2e
    float mk[4];
#pragma unroll
    for (int n = 0; n < 4; ++n) mk[n] = Mb[kv0 + n * 16 + ln] * LOG2E;
#pragma unroll
    for (int n = 0; n < 4; ++n)
#pragma unroll
      for (int r = 0; r < 8; ++r) s[n][r] = s[n][r] * SCALE_LOG2 + mk[n];

    // ---- online softmax (base-2): DPP row max, exp2, P spilled for transpose
#pragma unroll
    for (int r = 0; r < 8; ++r) {
      float v = fmaxf(fmaxf(s[0][r], s[1][r]), fmaxf(s[2][r], s[3][r]));
      v = rowmax16(v);
      const float mnew = fmaxf(mrow[r], v);
      const float al   = __builtin_amdgcn_exp2f(mrow[r] - mnew);
      mrow[r] = mnew;
      accL[r] *= al;
#pragma unroll
      for (int dt = 0; dt < 4; ++dt) acc[dt][r] *= al;
      const int prow = r + 8 * half;
#pragma unroll
      for (int n = 0; n < 4; ++n) {
        const float pv = __builtin_amdgcn_exp2f(s[n][r] - mnew);
        Plds[wave][prow * PPAD + n * 16 + ln] = f2bf(pv);
      }
    }
    // Same-wave LDS store->load ordering (CDNA5 split counters).
    asm volatile("s_wait_dscnt 0" ::: "memory");

    // ---- P as two A-matrix fragments (16x32 each, keys 0-31 / 32-63)
    v16bf pf[2];
#pragma unroll
    for (int ka = 0; ka < 2; ++ka) {
      const v8bf lo = *(const v8bf*)&Plds[wave][ln * PPAD + ka * 32 + half * 8];
      const v8bf hi = *(const v8bf*)&Plds[wave][ln * PPAD + ka * 32 + half * 8 + 16];
#pragma unroll
      for (int t = 0; t < 8; ++t) { pf[ka][t] = lo[t]; pf[ka][8 + t] = hi[t]; }
    }

    // ---- O += P * V  (4 d-tiles x 2 k-steps) and fused row-sum accumulate
#pragma unroll
    for (int dt = 0; dt < 4; ++dt) {
      const int d = dt * 16 + ln;
#pragma unroll
      for (int ka = 0; ka < 2; ++ka) {
        const v8bf lo = *(const v8bf*)&Vt[p][d * VPAD + ka * 32 + half * 8];
        const v8bf hi = *(const v8bf*)&Vt[p][d * VPAD + ka * 32 + half * 8 + 16];
        v16bf vb;
#pragma unroll
        for (int t = 0; t < 8; ++t) { vb[t] = lo[t]; vb[8 + t] = hi[t]; }
        acc[dt] = __builtin_amdgcn_wmma_f32_16x16x32_bf16(false, pf[ka], false, vb,
                                                          (short)0, acc[dt], false, false);
      }
    }
    accL = __builtin_amdgcn_wmma_f32_16x16x32_bf16(false, pf[0], false, ones,
                                                   (short)0, accL, false, false);
    accL = __builtin_amdgcn_wmma_f32_16x16x32_bf16(false, pf[1], false, ones,
                                                   (short)0, accL, false, false);

    // ---- convert + store NEXT tile into the other buffer; this VALU work
    //      co-executes with the O-GEMM WMMAs already in the XDL pipe.
    if (has_next) {
#pragma unroll
      for (int i = 0; i < 8; ++i) {
        const int key = skey + i * 8;
        v4bf kp;
#pragma unroll
        for (int t = 0; t < 4; ++t) kp[t] = f2bf(nkd[i][t]);
        *(v4bf*)&Klds[p ^ 1][key * KPAD + sd] = kp;
#pragma unroll
        for (int t = 0; t < 4; ++t) Vt[p ^ 1][(sd + t) * VPAD + key] = f2bf(nvd[i][t]);
      }
    }
    __syncthreads();
  }

  // ---- epilogue: normalize by the WMMA-accumulated row sums, store fp32
#pragma unroll
  for (int r = 0; r < 8; ++r) {
    const float inv = __builtin_amdgcn_rcpf(accL[r]);
    const int q = q0 + wave * 16 + r + 8 * half;
#pragma unroll
    for (int dt = 0; dt < 4; ++dt) {
      const int d = dt * 16 + ln;
      Ob[(size_t)q * ATT_D + d] = acc[dt][r] * inv;
    }
  }
}

extern "C" void kernel_launch(void* const* d_in, const int* in_sizes, int n_in,
                              void* d_out, int out_size, void* d_ws, size_t ws_size,
                              hipStream_t stream) {
  (void)in_sizes; (void)n_in; (void)out_size; (void)d_ws; (void)ws_size;
  const float* Q   = (const float*)d_in[0];
  const float* K   = (const float*)d_in[1];
  const float* V   = (const float*)d_in[2];
  const float* Msk = (const float*)d_in[3];
  float* O = (float*)d_out;
  dim3 grid(ATT_SQ / BLK_M, ATT_B);
  fa_fwd_gfx1250<<<grid, 128, 0, stream>>>(Q, K, V, Msk, O);
}
